// EncoderBlock_22067541966975
// MI455X (gfx1250) — compile-verified, matrix operand order checked by reading, not run on hardware
//
#include <hip/hip_runtime.h>

typedef __attribute__((ext_vector_type(16))) _Float16 v16h;
typedef __attribute__((ext_vector_type(8)))  _Float16 v8h;
typedef __attribute__((ext_vector_type(8)))  float    v8f;

#define BM 128
#define BN 64
#define BKK 32
#define LDT 40   // LDS row stride in halves (80B, 16B aligned)

__device__ __forceinline__ float gelu_f(float x) {
  return 0.5f * x * (1.0f + erff(x * 0.70710678118654752f));
}
__device__ __forceinline__ float sigmoid_f(float x) {
  return 1.0f / (1.0f + __expf(-x));
}

__device__ __forceinline__ v16h ldfrag(const _Float16* p) {
  v8h lo = *(const v8h*)p;
  v8h hi = *(const v8h*)(p + 16);
  v16h f;
  #pragma unroll
  for (int i = 0; i < 8; ++i) { f[i] = lo[i]; f[i + 8] = hi[i]; }
  return f;
}

// ---- staging macros: global fp32 -> registers -> f16 LDS (double buffered) ----
#define LOAD_A(K0) do {                                                        \
    if (aK) {                                                                  \
      const float* _s = Ab + (long long)(m0 + rA) * a_rs + (K0) + cA;          \
      *(float4*)(regA + 0)  = ((const float4*)_s)[0];                          \
      *(float4*)(regA + 4)  = ((const float4*)_s)[1];                          \
      *(float4*)(regA + 8)  = ((const float4*)_s)[2];                          \
      *(float4*)(regA + 12) = ((const float4*)_s)[3];                          \
    } else if (aM) {                                                           \
      const float* _s0 = Ab + (long long)((K0) + kA0) * a_cs + (m0 + mA0);     \
      *(float4*)(regA + 0) = ((const float4*)_s0)[0];                          \
      *(float4*)(regA + 4) = ((const float4*)_s0)[1];                          \
      const float* _s1 = Ab + (long long)((K0) + kA1) * a_cs + (m0 + mA0);     \
      *(float4*)(regA + 8)  = ((const float4*)_s1)[0];                         \
      *(float4*)(regA + 12) = ((const float4*)_s1)[1];                         \
    } else {                                                                   \
      const float* _s = Ab + (long long)(m0 + rA) * a_rs;                      \
      _Pragma("unroll")                                                        \
      for (int _j = 0; _j < 16; ++_j)                                          \
        regA[_j] = _s[(long long)((K0) + cA + _j) * a_cs];                     \
    }                                                                          \
  } while (0)

#define STORE_A(BI) do {                                                       \
    if (!aM) {                                                                 \
      v8h _h0, _h1;                                                            \
      _Pragma("unroll")                                                        \
      for (int _j = 0; _j < 8; ++_j) {                                         \
        _h0[_j] = (_Float16)regA[_j]; _h1[_j] = (_Float16)regA[_j + 8];        \
      }                                                                        \
      *(v8h*)&sA[BI][rA * LDT + cA]     = _h0;                                 \
      *(v8h*)&sA[BI][rA * LDT + cA + 8] = _h1;                                 \
    } else {                                                                   \
      _Pragma("unroll")                                                        \
      for (int _j = 0; _j < 8; ++_j)                                           \
        sA[BI][(mA0 + _j) * LDT + kA0] = (_Float16)regA[_j];                   \
      _Pragma("unroll")                                                        \
      for (int _j = 0; _j < 8; ++_j)                                           \
        sA[BI][(mA0 + _j) * LDT + kA1] = (_Float16)regA[_j + 8];               \
    }                                                                          \
  } while (0)

#define LOAD_B(K0) do {                                                        \
    if (bK) {                                                                  \
      if (nfull || (n0 + nB) < N) {                                            \
        const float* _s = Bb + (long long)(n0 + nB) * b_cs + (K0) + cB;        \
        *(float4*)(regB + 0) = ((const float4*)_s)[0];                         \
        *(float4*)(regB + 4) = ((const float4*)_s)[1];                         \
      } else {                                                                 \
        _Pragma("unroll") for (int _j = 0; _j < 8; ++_j) regB[_j] = 0.0f;      \
      }                                                                        \
    } else if (bN) {                                                           \
      const float* _s = Bb + (long long)((K0) + kB) * b_rs + (n0 + nlB);       \
      if (nfull || (n0 + nlB + 7) < N) {                                       \
        *(float4*)(regB + 0) = ((const float4*)_s)[0];                         \
        *(float4*)(regB + 4) = ((const float4*)_s)[1];                         \
      } else {                                                                 \
        _Pragma("unroll")                                                      \
        for (int _j = 0; _j < 8; ++_j)                                         \
          regB[_j] = ((n0 + nlB + _j) < N) ? _s[_j] : 0.0f;                    \
      }                                                                        \
    } else {                                                                   \
      _Pragma("unroll")                                                        \
      for (int _j = 0; _j < 8; ++_j) {                                         \
        float _v = 0.0f;                                                       \
        if (nfull || (n0 + nB) < N)                                            \
          _v = Bb[(long long)((K0) + cB + _j) * b_rs + (long long)(n0 + nB) * b_cs]; \
        regB[_j] = _v;                                                         \
      }                                                                        \
    }                                                                          \
  } while (0)

#define STORE_B(BI) do {                                                       \
    v8h _h;                                                                    \
    _Pragma("unroll")                                                          \
    for (int _j = 0; _j < 8; ++_j) _h[_j] = (_Float16)regB[_j];                \
    if (bN) {                                                                  \
      _Pragma("unroll")                                                        \
      for (int _j = 0; _j < 8; ++_j) sB[BI][(nlB + _j) * LDT + kB] = _h[_j];   \
    } else {                                                                   \
      *(v8h*)&sB[BI][nB * LDT + cB] = _h;                                      \
    }                                                                          \
  } while (0)

// ---------------------------------------------------------------------------
// Strided-batched GEMM, D[z](m,n) = epi( sum_k A[z](m,k) * B[z](k,n) )
// Contract: M % 128 == 0, K % 32 == 0 (N arbitrary, guarded).
// Software-pipelined: next K-tile global->regs issued before WMMAs,
// regs->LDS after WMMAs, one barrier per K-step, double-buffered LDS.
// ---------------------------------------------------------------------------
__global__ void __launch_bounds__(256) gemm_wmma_f16(
    const float* __restrict__ A, long long a_rs, long long a_cs, long long a_bs,
    const float* __restrict__ B, long long b_rs, long long b_cs, long long b_bs,
    float* __restrict__ D, long long d_rs, long long d_cs, long long d_bs,
    const float* __restrict__ bias,
    const float* __restrict__ mul, long long m_rs, long long m_cs, long long m_bs,
    const float* __restrict__ alpha_ptr, float alpha,
    int act, int accum, int M, int N, int K)
{
  __shared__ __align__(16) _Float16 sA[2][BM * LDT];
  __shared__ __align__(16) _Float16 sB[2][BN * LDT];

  const int m0 = blockIdx.x * BM;
  const int n0 = blockIdx.y * BN;
  const int z  = blockIdx.z;
  const float* Ab = A + (long long)z * a_bs;
  const float* Bb = B + (long long)z * b_bs;

  const int tid  = threadIdx.x;
  const int lane = tid & 31;
  const int w    = tid >> 5;
  const int wm   = w >> 1;        // 0..3 -> 32-row slab
  const int wn   = w & 1;         // 0..1 -> 32-col slab
  const int kb   = (lane >> 4) * 8;
  const int r16  = lane & 15;

  const bool aK = (a_cs == 1);
  const bool aM = (a_rs == 1);
  const bool bK = (b_rs == 1);
  const bool bN = (b_cs == 1);
  const bool nfull = (n0 + BN <= N);

  // invariant per-thread staging coordinates
  const int rA  = tid >> 1,  cA  = (tid & 1) * 16;           // k-contig / scalar A
  const int kA0 = tid >> 4,  mA0 = (tid & 15) * 8;           // m-contig A group 0
  const int kA1 = kA0 + 16;                                  // m-contig A group 1
  const int nB  = tid >> 2,  cB  = (tid & 3) * 8;            // k-contig / scalar B
  const int kB  = tid >> 3,  nlB = (tid & 7) * 8;            // n-contig B

  v8f acc00 = {}, acc01 = {}, acc10 = {}, acc11 = {};
  float regA[16];
  float regB[8];

  // prologue: stage tile 0
  LOAD_A(0); LOAD_B(0);
  STORE_A(0); STORE_B(0);
  __syncthreads();

  int cur = 0;
  for (int k0 = 0; k0 < K; k0 += BKK) {
    const int kn = k0 + BKK;
    const bool hn = kn < K;
    if (hn) { LOAD_A(kn); LOAD_B(kn); }     // global -> regs (overlaps WMMAs)
    if (kn + BKK < K) {                      // speculative prefetch, tile +2
      __builtin_prefetch(Ab + (long long)(m0 + rA) * a_rs + (long long)(kn + BKK) * a_cs, 0, 1);
      __builtin_prefetch(Bb + (long long)(kn + BKK) * b_rs + (long long)(n0 + nB) * b_cs, 0, 1);
    }

    const _Float16* pSA = sA[cur];
    const _Float16* pSB = sB[cur];
    v16h af0 = ldfrag(&pSA[(wm * 32 +      r16) * LDT + kb]);
    v16h af1 = ldfrag(&pSA[(wm * 32 + 16 + r16) * LDT + kb]);
    v16h bf0 = ldfrag(&pSB[(wn * 32 +      r16) * LDT + kb]);
    v16h bf1 = ldfrag(&pSB[(wn * 32 + 16 + r16) * LDT + kb]);

    acc00 = __builtin_amdgcn_wmma_f32_16x16x32_f16(false, af0, false, bf0, (short)0, acc00, false, false);
    acc01 = __builtin_amdgcn_wmma_f32_16x16x32_f16(false, af0, false, bf1, (short)0, acc01, false, false);
    acc10 = __builtin_amdgcn_wmma_f32_16x16x32_f16(false, af1, false, bf0, (short)0, acc10, false, false);
    acc11 = __builtin_amdgcn_wmma_f32_16x16x32_f16(false, af1, false, bf1, (short)0, acc11, false, false);

    if (hn) { STORE_A(cur ^ 1); STORE_B(cur ^ 1); }   // regs -> other LDS buffer
    __syncthreads();
    cur ^= 1;
  }

  float asc = alpha;
  if (alpha_ptr) asc *= alpha_ptr[0];
  const float* Mb = mul ? (mul + (long long)z * m_bs) : (const float*)0;
  float* Db = D + (long long)z * d_bs;
  const int mhi = 8 * (lane >> 4);

  #pragma unroll
  for (int tn = 0; tn < 2; ++tn) {
    int gn = n0 + wn * 32 + tn * 16 + r16;
    bool nok = gn < N;
    float bsv = (bias && nok) ? bias[gn] : 0.0f;
    #pragma unroll
    for (int tm = 0; tm < 2; ++tm) {
      v8f av = (tm == 0) ? ((tn == 0) ? acc00 : acc01) : ((tn == 0) ? acc10 : acc11);
      int gmb = m0 + wm * 32 + tm * 16 + mhi;
      if (nok) {
        #pragma unroll
        for (int r = 0; r < 8; ++r) {
          int gm = gmb + r;
          float v = (av[r] + bsv) * asc;
          if (act == 1)      v = gelu_f(v);
          else if (act == 2) v = sigmoid_f(v);
          if (Mb) v *= Mb[(long long)gm * m_rs + (long long)gn * m_cs];
          long long off = (long long)gm * d_rs + (long long)gn * d_cs;
          if (accum) v += Db[off];
          Db[off] = v;
        }
      }
    }
  }
}

// ---------------------------------------------------------------------------
// LayerNorm over last dim d; one row per block; in-place safe.
// ---------------------------------------------------------------------------
__global__ void layernorm_kernel(const float* __restrict__ x,
                                 const float* __restrict__ g,
                                 const float* __restrict__ b,
                                 float* __restrict__ out, int d)
{
  long long row = blockIdx.x;
  const float* xr = x + row * (long long)d;
  float s = 0.f, s2 = 0.f;
  for (int i = threadIdx.x; i < d; i += blockDim.x) {
    float v = xr[i]; s += v; s2 += v * v;
  }
  __shared__ float rs[8], rs2[8];
  for (int o = 16; o > 0; o >>= 1) { s += __shfl_down(s, o); s2 += __shfl_down(s2, o); }
  int wv = threadIdx.x >> 5, ln = threadIdx.x & 31;
  if (ln == 0) { rs[wv] = s; rs2[wv] = s2; }
  __syncthreads();
  int nw = blockDim.x >> 5;
  if (threadIdx.x == 0) {
    float a = 0.f, a2 = 0.f;
    for (int i = 0; i < nw; ++i) { a += rs[i]; a2 += rs2[i]; }
    rs[0] = a; rs2[0] = a2;
  }
  __syncthreads();
  float mu  = rs[0] / (float)d;
  float var = rs2[0] / (float)d - mu * mu;
  float inv = rsqrtf(var + 1e-5f);
  for (int i = threadIdx.x; i < d; i += blockDim.x) {
    out[row * (long long)d + i] = (xr[i] - mu) * inv * g[i] + b[i];
  }
}

// softmax over j of logits[h,i,j] + pb[i,j,h]; blockIdx.x = h*512+i
__global__ void softmax_bias_kernel(float* __restrict__ logits,
                                    const float* __restrict__ pb)
{
  int h = blockIdx.x >> 9;
  int i = blockIdx.x & 511;
  float* row = logits + (long long)blockIdx.x * 512;
  const float* pbr = pb + (long long)i * 512 * 8 + h;
  int tid = threadIdx.x;
  float v0 = row[tid]       + pbr[(long long)tid * 8];
  float v1 = row[tid + 256] + pbr[(long long)(tid + 256) * 8];

  __shared__ float rmax[8], rsum[8];
  float m = fmaxf(v0, v1);
  for (int o = 16; o > 0; o >>= 1) m = fmaxf(m, __shfl_down(m, o));
  int wv = tid >> 5, ln = tid & 31;
  if (ln == 0) rmax[wv] = m;
  __syncthreads();
  if (tid == 0) {
    float mm = rmax[0];
    for (int k = 1; k < 8; ++k) mm = fmaxf(mm, rmax[k]);
    rmax[0] = mm;
  }
  __syncthreads();
  float mx = rmax[0];
  float e0 = __expf(v0 - mx), e1 = __expf(v1 - mx);
  float s = e0 + e1;
  for (int o = 16; o > 0; o >>= 1) s += __shfl_down(s, o);
  if (ln == 0) rsum[wv] = s;
  __syncthreads();
  if (tid == 0) {
    float ss = 0.f;
    for (int k = 0; k < 8; ++k) ss += rsum[k];
    rsum[0] = ss;
  }
  __syncthreads();
  float inv = 1.0f / rsum[0];
  row[tid]       = e0 * inv;
  row[tid + 256] = e1 * inv;
}

// pb[ij,h] = sum_c pair[ij,c] * W[c,h]
__global__ void pair_bias_kernel(const float* __restrict__ pair,
                                 const float* __restrict__ wgt,
                                 float* __restrict__ pb)
{
  __shared__ float sw[1024];
  for (int i = threadIdx.x; i < 1024; i += blockDim.x) sw[i] = wgt[i];
  __syncthreads();
  long long idx = (long long)blockIdx.x * blockDim.x + threadIdx.x;
  long long ij = idx >> 3;
  int h = (int)(idx & 7);
  const float* pr = pair + ij * 128;
  float acc = 0.f;
  for (int c = 0; c < 128; ++c) acc += pr[c] * sw[c * 8 + h];
  pb[idx] = acc;
}

// sbuf[0] = rsqrt(max(sum(atom_mask>0),1)); sbuf[1+i] = mask[i]
__global__ void mask_scale_kernel(const float* __restrict__ am,
                                  float* __restrict__ sbuf, int n)
{
  __shared__ float red[8];
  float s = 0.f;
  for (int i = threadIdx.x; i < n; i += blockDim.x) {
    float mv = (am[i] > 0.f) ? 1.0f : 0.0f;
    sbuf[1 + i] = mv;
    s += mv;
  }
  for (int o = 16; o > 0; o >>= 1) s += __shfl_down(s, o);
  int wv = threadIdx.x >> 5, ln = threadIdx.x & 31;
  if (ln == 0) red[wv] = s;
  __syncthreads();
  if (threadIdx.x == 0) {
    float t = 0.f;
    for (int k = 0; k < 8; ++k) t += red[k];
    sbuf[0] = rsqrtf(fmaxf(t, 1.0f));
  }
}

// fused: v(ij,c) = abp*sg*pair_mask[ij]*scale, transposed out to planes
// outT[c * nij + ij] for c in 0..63 (a-planes 0..31, b-planes 32..63)
__global__ void __launch_bounds__(256) combine_transpose_kernel(
    const float* __restrict__ abp, const float* __restrict__ sg,
    const float* __restrict__ pm, const float* __restrict__ sptr,
    float* __restrict__ outT, long long nij)
{
  __shared__ float tile[64][65];
  float sc = sptr[0];
  long long ij0 = (long long)blockIdx.x * 64;
  #pragma unroll
  for (int i = 0; i < 16; ++i) {
    int e = threadIdx.x + 256 * i;
    int r = e >> 6, c = e & 63;
    long long ij = ij0 + r;
    long long idx = ij * 64 + c;
    tile[r][c] = abp[idx] * sg[idx] * pm[ij] * sc;
  }
  __syncthreads();
  #pragma unroll
  for (int i = 0; i < 16; ++i) {
    int e = threadIdx.x + 256 * i;
    int c = e >> 6, r = e & 63;
    outT[(long long)c * nij + ij0 + r] = tile[r][c];
  }
}

// ---------------------------------------------------------------------------
// host side
// ---------------------------------------------------------------------------
static inline void run_gemm(hipStream_t st,
    const float* A, long long ars, long long acs, long long abs_,
    const float* B, long long brs, long long bcs, long long bbs,
    float* D, long long drs, long long dcs, long long dbs,
    const float* bias,
    const float* mul, long long mrs, long long mcs, long long mbs,
    const float* aptr, float alpha, int act, int accum,
    int M, int N, int K, int batches)
{
  dim3 grid((M + BM - 1) / BM, (N + BN - 1) / BN, batches);
  gemm_wmma_f16<<<grid, 256, 0, st>>>(A, ars, acs, abs_, B, brs, bcs, bbs,
                                      D, drs, dcs, dbs, bias,
                                      mul, mrs, mcs, mbs,
                                      aptr, alpha, act, accum, M, N, K);
}

struct LayerP {
  const float *gate_b,*gate_w,*k_b,*k_w,*out_b,*out_w,*pb_w,*q_b,*q_w,*v_b,*v_w;
  const float *fc1_b,*fc1_w,*fc2_b,*fc2_w;
  const float *ln_attn_b,*ln_attn_g,*ln_final_b,*ln_final_g,*ln_opm_b,*ln_opm_g;
  const float *ln_pffn_b,*ln_pffn_g,*ln_tri_b,*ln_tri_g;
  const float *li_b,*li_w,*lo_b,*lo_w;
  const float *pf1_b,*pf1_w,*pf2_b,*pf2_w;
  const float *abg_b,*abg_w,*abp_b,*abp_w,*tg_b,*tg_w,*tln_b,*tln_g,*tz_b,*tz_w;
};

extern "C" void kernel_launch(void* const* d_in, const int* in_sizes, int n_in,
                              void* d_out, int out_size, void* d_ws, size_t ws_size,
                              hipStream_t stream) {
  (void)in_sizes; (void)n_in; (void)out_size; (void)ws_size;
  const int Nn = 512, E = 768, H = 8, HD = 96, P = 128, F = 3072, DH = 32, LL = 6;
  const long long NE = (long long)Nn * E;
  const long long NN = (long long)Nn * Nn;
  const long long NNP = NN * P;

  int idx = 0;
  auto FP = [&]() { return (const float*)d_in[idx++]; };
  const float* node_in   = FP();
  const float* pair_in   = FP();
  const float* atom_mask = FP();
  const float* pair_mask = FP();

  // params flattened in JAX sorted-key pytree order: layers, ln_node, ln_pair
  LayerP L[6];
  for (int l = 0; l < LL; ++l) {
    LayerP& p = L[l];
    p.gate_b = FP(); p.gate_w = FP();
    p.k_b    = FP(); p.k_w    = FP();
    p.out_b  = FP(); p.out_w  = FP();
    p.pb_w   = FP();
    p.q_b    = FP(); p.q_w    = FP();
    p.v_b    = FP(); p.v_w    = FP();
    p.fc1_b = FP(); p.fc1_w = FP();
    p.fc2_b = FP(); p.fc2_w = FP();
    p.ln_attn_b  = FP(); p.ln_attn_g  = FP();
    p.ln_final_b = FP(); p.ln_final_g = FP();
    p.ln_opm_b   = FP(); p.ln_opm_g   = FP();
    p.ln_pffn_b  = FP(); p.ln_pffn_g  = FP();
    p.ln_tri_b   = FP(); p.ln_tri_g   = FP();
    p.li_b = FP(); p.li_w = FP();
    p.lo_b = FP(); p.lo_w = FP();
    p.pf1_b = FP(); p.pf1_w = FP();
    p.pf2_b = FP(); p.pf2_w = FP();
    p.abg_b = FP(); p.abg_w = FP();
    p.abp_b = FP(); p.abp_w = FP();
    p.tg_b  = FP(); p.tg_w  = FP();
    p.tln_b = FP(); p.tln_g = FP();
    p.tz_b  = FP(); p.tz_w  = FP();
  }
  const float* ln_node_b = FP(); const float* ln_node_g = FP();
  const float* ln_pair_b = FP(); const float* ln_pair_g = FP();

  // workspace carve-up (fp32)
  float* ws = (float*)d_ws;
  size_t off = 0;
  auto alloc = [&](size_t n) { float* p = ws + off; off += n; return p; };
  float* sbuf = alloc(1024);
  float* node = alloc(NE);
  float* xln  = alloc(NE);
  float* qb   = alloc(NE);
  float* kb   = alloc(NE);
  float* vb   = alloc(NE);
  float* ob   = alloc(NE);
  float* og   = alloc(NE);
  float* h1   = alloc((size_t)Nn * F);
  float* pbb  = alloc(NN * H);
  float* lg   = alloc((size_t)H * NN);
  float* abo  = alloc((size_t)Nn * 2 * DH);
  float* tb   = alloc((size_t)Nn * DH * P);
  float* pair = alloc(NNP);
  float* P1   = alloc(NNP);      // pair LN  / transposed gated a,b planes
  float* P2   = alloc(NNP);      // trimul g / pair-ffn hidden
  float* Bu1  = alloc(NN * 2 * DH);
  float* Bu2  = alloc(NN * 2 * DH);

  const float alpha_q = 0.10206207261596577f;   // HD^-0.5

  mask_scale_kernel<<<1, 256, 0, stream>>>(atom_mask, sbuf, Nn);
  layernorm_kernel<<<Nn, 256, 0, stream>>>(node_in, ln_node_g, ln_node_b, node, E);
  layernorm_kernel<<<(unsigned)NN, 128, 0, stream>>>(pair_in, ln_pair_g, ln_pair_b, pair, P);

  for (int l = 0; l < LL; ++l) {
    const LayerP& p = L[l];

    // ---------------- attention ----------------
    layernorm_kernel<<<Nn, 256, 0, stream>>>(node, p.ln_attn_g, p.ln_attn_b, xln, E);
    run_gemm(stream, xln, E,1,0, p.q_w, E,1,0, qb, E,1,0, p.q_b, 0,0,0,0, 0, alpha_q, 0,0, Nn,E,E, 1);
    run_gemm(stream, xln, E,1,0, p.k_w, E,1,0, kb, E,1,0, p.k_b, 0,0,0,0, 0, 1.0f, 0,0, Nn,E,E, 1);
    run_gemm(stream, xln, E,1,0, p.v_w, E,1,0, vb, E,1,0, p.v_b, 0,0,0,0, 0, 1.0f, 0,0, Nn,E,E, 1);
    pair_bias_kernel<<<(unsigned)(NN * H / 256), 256, 0, stream>>>(pair, p.pb_w, pbb);
    // logits[h,i,j] = q_h K_h^T   (B is k-contiguous)
    run_gemm(stream, qb, E,1,HD,  kb, 1,E,HD,  lg, Nn,1,NN, 0, 0,0,0,0, 0, 1.0f, 0,0, Nn,Nn,HD, H);
    softmax_bias_kernel<<<(unsigned)(H * Nn), 256, 0, stream>>>(lg, pbb);
    // o = A_h V_h   (N=96 guarded)
    run_gemm(stream, lg, Nn,1,NN,  vb, E,1,HD,  ob, E,1,HD, 0, 0,0,0,0, 0, 1.0f, 0,0, Nn,HD,Nn, H);
    run_gemm(stream, xln, E,1,0, p.gate_w, E,1,0, og, E,1,0, p.gate_b, ob, E,1,0, 0, 1.0f, 2,0, Nn,E,E, 1);
    run_gemm(stream, og, E,1,0, p.out_w, E,1,0, node, E,1,0, p.out_b, 0,0,0,0, 0, 1.0f, 0,1, Nn,E,E, 1);

    // ---------------- node FFN ----------------
    layernorm_kernel<<<Nn, 256, 0, stream>>>(node, p.ln_final_g, p.ln_final_b, xln, E);
    run_gemm(stream, xln, E,1,0, p.fc1_w, F,1,0, h1, F,1,0, p.fc1_b, 0,0,0,0, 0, 1.0f, 1,0, Nn,F,E, 1);
    run_gemm(stream, h1, F,1,0, p.fc2_w, E,1,0, node, E,1,0, p.fc2_b, 0,0,0,0, 0, 1.0f, 0,1, Nn,E,F, 1);

    // ---------------- outer product mean ----------------
    layernorm_kernel<<<Nn, 256, 0, stream>>>(node, p.ln_opm_g, p.ln_opm_b, xln, E);
    run_gemm(stream, xln, E,1,0, p.li_w, 2*DH,1,0, abo, 2*DH,1,0, p.li_b,
             sbuf + 1, 1,0,0, 0, 1.0f, 0,0, Nn,2*DH,E, 1);
    // t[j,x,p] (batch over x)
    run_gemm(stream, abo + DH, 2*DH,1,0,  p.lo_w, P,1,(long long)DH*P,
             tb, (long long)DH*P,1,P, 0, 0,0,0,0, 0, 1.0f, 0,0, Nn,P,DH, DH);
    // pair[i,j,:] += (sum_x a[i,x] t[j,x,:] + b) * op_norm  (batch over j)
    run_gemm(stream, abo, 2*DH,1,0,  tb, P,1,(long long)DH*P,
             pair, (long long)Nn*P,1,P, p.lo_b, 0,0,0,0, sbuf, 1.0f, 0,1, Nn,P,DH, Nn);

    // ---------------- triangle multiplication ----------------
    layernorm_kernel<<<(unsigned)NN, 128, 0, stream>>>(pair, p.ln_tri_g, p.ln_tri_b, P1, P);
    run_gemm(stream, P1, P,1,0, p.tg_w, P,1,0, P2, P,1,0, p.tg_b, 0,0,0,0, 0, 1.0f, 2,0, (int)NN,P,P, 1);
    run_gemm(stream, P1, P,1,0, p.abp_w, 2*DH,1,0, Bu1, 2*DH,1,0, p.abp_b, 0,0,0,0, 0, 1.0f, 0,0, (int)NN,2*DH,P, 1);
    run_gemm(stream, P1, P,1,0, p.abg_w, 2*DH,1,0, Bu2, 2*DH,1,0, p.abg_b, 0,0,0,0, 0, 1.0f, 2,0, (int)NN,2*DH,P, 1);
    // fused gate/mask/scale + transpose to per-channel planes in P1
    combine_transpose_kernel<<<(unsigned)(NN / 64), 256, 0, stream>>>(Bu1, Bu2, pair_mask, sbuf, P1, NN);
    // x_out[i,j,c] = sum_k aT[c][i*512+k] * bT[c][j*512+k]   (batch over c)
    run_gemm(stream, P1, Nn,1,NN,  P1 + 32*NN, 1,Nn,NN,
             Bu2, (long long)Nn*2*DH, 2*DH, 1, 0, 0,0,0,0, 0, 1.0f, 0,0, Nn,Nn,Nn, DH);
    // x_in[i,j,c] = sum_k aT[c][k*512+i] * bT[c][k*512+j]    (batch over c)
    run_gemm(stream, P1, 1,Nn,NN,  P1 + 32*NN, Nn,1,NN,
             Bu2 + DH, (long long)Nn*2*DH, 2*DH, 1, 0, 0,0,0,0, 0, 1.0f, 0,0, Nn,Nn,Nn, DH);
    layernorm_kernel<<<(unsigned)NN, 64, 0, stream>>>(Bu2, p.tln_g, p.tln_b, Bu2, 2*DH);
    run_gemm(stream, Bu2, 2*DH,1,0, p.tz_w, P,1,0, pair, P,1,0, p.tz_b,
             P2, P,1,0, 0, 1.0f, 0,1, (int)NN,P,2*DH, 1);

    // ---------------- pair FFN ----------------
    layernorm_kernel<<<(unsigned)NN, 128, 0, stream>>>(pair, p.ln_pffn_g, p.ln_pffn_b, P1, P);
    run_gemm(stream, P1, P,1,0, p.pf1_w, P,1,0, P2, P,1,0, p.pf1_b, 0,0,0,0, 0, 1.0f, 1,0, (int)NN,P,P, 1);
    run_gemm(stream, P2, P,1,0, p.pf2_w, P,1,0, pair, P,1,0, p.pf2_b, 0,0,0,0, 0, 1.0f, 0,1, (int)NN,P,P, 1);
  }

  hipMemcpyAsync(d_out, node, NE * sizeof(float), hipMemcpyDeviceToDevice, stream);
  hipMemcpyAsync((float*)d_out + NE, pair, NNP * sizeof(float), hipMemcpyDeviceToDevice, stream);
}
